// Head_309237646138
// MI455X (gfx1250) — matrix operand order repaired and checked by reading, not compiled, not run on hardware
//
#include <hip/hip_runtime.h>

#define Bsz  4
#define Tseq 4096
#define Cdim 1024
#define HSd  64

typedef __attribute__((ext_vector_type(16))) __bf16 v16bf;
typedef __attribute__((ext_vector_type(8)))  __bf16 v8bf;
typedef __attribute__((ext_vector_type(8)))  float  v8f;

union V16U { v16bf v; v8bf h[2]; };

static __device__ __forceinline__ v16bf mk16(v8bf lo, v8bf hi) {
  V16U u; u.h[0] = lo; u.h[1] = hi; return u.v;
}

static __device__ __forceinline__ v8f wmma_bf16(v16bf a, v16bf b, v8f c) {
  // 8 args: (neg_a, A, neg_b, B, c_mod, C, reuse_a, reuse_b)
  return __builtin_amdgcn_wmma_f32_16x16x32_bf16(false, a, false, b, (short)0, c, false, false);
}

// ---------------------------------------------------------------------------
// Kernel 1: transpose Wq/Wk/Wv ([C][HS] f32) -> wt[mat][HS][C] bf16.
// Softmax scale 1/sqrt(HS)=0.125 folded into Wq.
// ---------------------------------------------------------------------------
__global__ __launch_bounds__(256) void wt_transpose_kernel(
    const float* __restrict__ Wq, const float* __restrict__ Wk,
    const float* __restrict__ Wv, __bf16* __restrict__ wt) {
  int idx = blockIdx.x * 256 + threadIdx.x;          // 3*HS*C total threads
  int mat = idx / (HSd * Cdim);
  int rem = idx - mat * (HSd * Cdim);
  int h = rem / Cdim;
  int c = rem - h * Cdim;
  const float* W = (mat == 0) ? Wq : ((mat == 1) ? Wk : Wv);
  float s = (mat == 0) ? 0.125f : 1.0f;
  wt[idx] = (__bf16)(W[c * HSd + h] * s);
}

// ---------------------------------------------------------------------------
// Kernel 2: QKV projection. Block = 3 waves (q/k/v), one 16-token tile each.
// B fragments batched ahead of the 4 WMMAs so the matrix pipe streams.
// Writes Q,K row-major bf16 [B*T][HS] and V transposed bf16 [B][HS][T].
// ---------------------------------------------------------------------------
__global__ __launch_bounds__(96) void qkv_proj_kernel(
    const float* __restrict__ x, const __bf16* __restrict__ wt,
    __bf16* __restrict__ qb, __bf16* __restrict__ kb, __bf16* __restrict__ vt) {
  const int lane = threadIdx.x & 31;
  const int mat  = threadIdx.x >> 5;                 // 0=q, 1=k, 2=v
  const int col  = lane & 15;
  const int hi   = lane >> 4;
  const int row0 = blockIdx.x * 16;                  // token tile base
  const float* xr = x + (size_t)(row0 + col) * Cdim; // this lane's A row
  const __bf16* wmat = wt + (size_t)mat * HSd * Cdim;

  v8f acc[4] = {};
  for (int c0 = 0; c0 < Cdim; c0 += 32) {
    // A fragment: K = c0 + 16*hi + [0,16), f32->bf16
    v8f f0 = *(const v8f*)(xr + c0 + 16 * hi);
    v8f f1 = *(const v8f*)(xr + c0 + 16 * hi + 8);
    // B fragments for all 4 HS tiles, issued before any WMMA
    v16bf bfr[4];
#pragma unroll
    for (int nt = 0; nt < 4; ++nt)
      bfr[nt] = *(const v16bf*)(wmat + (size_t)(nt * 16 + col) * Cdim + c0 + 16 * hi);
    if (c0 + 32 < Cdim)
      __builtin_prefetch(xr + c0 + 32, 0, 3);        // next A chunk -> WGP$
    V16U ua;
#pragma unroll
    for (int e = 0; e < 8; ++e) { ua.h[0][e] = (__bf16)f0[e]; ua.h[1][e] = (__bf16)f1[e]; }
    v16bf a = ua.v;
#pragma unroll
    for (int nt = 0; nt < 4; ++nt)
      acc[nt] = wmma_bf16(a, bfr[nt], acc[nt]);
  }
#pragma unroll
  for (int nt = 0; nt < 4; ++nt) {
#pragma unroll
    for (int r = 0; r < 8; ++r) {
      int tok = row0 + r + 8 * hi;                   // D row = r + 8*hi
      int h   = nt * 16 + col;                       // D col = lane&15
      __bf16 val = (__bf16)acc[nt][r];
      if (mat == 0)      qb[(size_t)tok * HSd + h] = val;
      else if (mat == 1) kb[(size_t)tok * HSd + h] = val;
      else {
        int bidx = tok >> 12;                        // / Tseq
        int t    = tok & (Tseq - 1);
        vt[((size_t)bidx * HSd + h) * Tseq + t] = val;
      }
    }
  }
}

// ---------------------------------------------------------------------------
// Kernel 3: causal flash attention. Block = 4 waves; wave owns a 16-query
// tile; loop over 32-key blocks up to the diagonal. K fragments batched
// before the S WMMAs; V fragments issued before the softmax VALU block so
// their latency hides under ~200 VALU ops; P staged through LDS
// (D-layout -> A-layout); O += P*V. fp32 output.
// ---------------------------------------------------------------------------
__global__ __launch_bounds__(128) void flash_attn_kernel(
    const __bf16* __restrict__ qb, const __bf16* __restrict__ kb,
    const __bf16* __restrict__ vt, float* __restrict__ out) {
  __shared__ __bf16 plds[4][16 * 32];                // 1KB per wave
  const int lane = threadIdx.x & 31;
  const int wv   = threadIdx.x >> 5;
  const int col  = lane & 15;
  const int hi   = lane >> 4;
  const int b    = blockIdx.x >> 6;                  // Tseq/64 = 64 tiles/batch
  const int tile = blockIdx.x & 63;
  const int q0   = tile * 64 + wv * 16;

  // Q A-fragments for the two 32-wide K chunks of HS
  const __bf16* qrow = qb + (size_t)(b * Tseq + q0 + col) * HSd;
  v16bf aq0 = mk16(*(const v8bf*)(qrow + 16 * hi),      *(const v8bf*)(qrow + 8 + 16 * hi));
  v16bf aq1 = mk16(*(const v8bf*)(qrow + 32 + 16 * hi), *(const v8bf*)(qrow + 40 + 16 * hi));

  const __bf16* kbase = kb + (size_t)b * Tseq * HSd;
  const __bf16* vbase = vt + (size_t)b * HSd * Tseq;

  v8f o[4] = {};
  float mi[8], li[8];
#pragma unroll
  for (int r = 0; r < 8; ++r) { mi[r] = -__builtin_inff(); li[r] = 0.0f; }

  __bf16* pw = &plds[wv][0];
  const int qlast = q0 + 15;
  for (int kbk = 0; kbk <= qlast; kbk += 32) {
    // --- K fragments (all 4), issued before any WMMA ---------------------
    const __bf16* k0 = kbase + (size_t)(kbk + col) * HSd + 16 * hi;
    const __bf16* k1 = kbase + (size_t)(kbk + 16 + col) * HSd + 16 * hi;
    v16bf kf00 = *(const v16bf*)(k0);
    v16bf kf01 = *(const v16bf*)(k0 + 32);
    v16bf kf10 = *(const v16bf*)(k1);
    v16bf kf11 = *(const v16bf*)(k1 + 32);
    if (kbk + 32 <= qlast)                           // next key block -> WGP$
      __builtin_prefetch(k0 + (size_t)32 * HSd, 0, 3);

    v8f s0 = {}, s1 = {};
    s0 = wmma_bf16(aq0, kf00, s0);
    s0 = wmma_bf16(aq1, kf01, s0);
    s1 = wmma_bf16(aq0, kf10, s1);
    s1 = wmma_bf16(aq1, kf11, s1);

    // --- V fragments issued now; latency hides under the softmax VALU ----
    v16bf vf[4];
#pragma unroll
    for (int nt = 0; nt < 4; ++nt)
      vf[nt] = *(const v16bf*)(vbase + (size_t)(nt * 16 + col) * Tseq + kbk + 16 * hi);

    const bool needMask = (kbk + 31 > q0);
#pragma unroll
    for (int r = 0; r < 8; ++r) {
      const int qg = q0 + r + 8 * hi;                // this register's query row
      float a0 = s0[r], a1 = s1[r];
      if (needMask) {
        if (kbk + col > qg)      a0 = -1e30f;
        if (kbk + 16 + col > qg) a1 = -1e30f;
      }
      float mx = fmaxf(a0, a1);
      mx = fmaxf(mx, __shfl_xor(mx, 1, 32));
      mx = fmaxf(mx, __shfl_xor(mx, 2, 32));
      mx = fmaxf(mx, __shfl_xor(mx, 4, 32));
      mx = fmaxf(mx, __shfl_xor(mx, 8, 32));
      float mnew  = fmaxf(mi[r], mx);
      float alpha = __expf(mi[r] - mnew);
      float p0 = __expf(a0 - mnew);
      float p1 = __expf(a1 - mnew);
      float rs = p0 + p1;
      rs += __shfl_xor(rs, 1, 32);
      rs += __shfl_xor(rs, 2, 32);
      rs += __shfl_xor(rs, 4, 32);
      rs += __shfl_xor(rs, 8, 32);
      li[r] = li[r] * alpha + rs;
      mi[r] = mnew;
#pragma unroll
      for (int nt = 0; nt < 4; ++nt) o[nt][r] *= alpha;
      pw[(r + 8 * hi) * 32 + col]      = (__bf16)p0;
      pw[(r + 8 * hi) * 32 + 16 + col] = (__bf16)p1;
    }
    // cross-lane LDS handoff within the wave (DS ops are in-order per wave,
    // but fence the compiler and the counter explicitly)
    asm volatile("s_wait_dscnt 0" ::: "memory");
    v16bf ap = mk16(*(const v8bf*)(pw + col * 32 + 16 * hi),
                    *(const v8bf*)(pw + col * 32 + 8 + 16 * hi));
#pragma unroll
    for (int nt = 0; nt < 4; ++nt)
      o[nt] = wmma_bf16(ap, vf[nt], o[nt]);
  }
#pragma unroll
  for (int r = 0; r < 8; ++r) {
    float inv = 1.0f / li[r];
    int qg = q0 + r + 8 * hi;
    float* orow = out + (size_t)(b * Tseq + qg) * HSd;
#pragma unroll
    for (int nt = 0; nt < 4; ++nt) orow[nt * 16 + col] = o[nt][r] * inv;
  }
}

// ---------------------------------------------------------------------------
extern "C" void kernel_launch(void* const* d_in, const int* in_sizes, int n_in,
                              void* d_out, int out_size, void* d_ws, size_t ws_size,
                              hipStream_t stream) {
  const float* x  = (const float*)d_in[0];
  const float* Wq = (const float*)d_in[1];
  const float* Wk = (const float*)d_in[2];
  const float* Wv = (const float*)d_in[3];
  float* out = (float*)d_out;

  // workspace layout (bf16): wt[3][HS][C], q[B*T][HS], k[B*T][HS], vt[B][HS][T]
  const size_t WT_BYTES = (size_t)3 * HSd * Cdim * sizeof(__bf16);   // 384 KB
  const size_t QK_BYTES = (size_t)Bsz * Tseq * HSd * sizeof(__bf16); // 2 MB each
  char* ws = (char*)d_ws;
  __bf16* wt = (__bf16*)ws;
  __bf16* qb = (__bf16*)(ws + WT_BYTES);
  __bf16* kb = (__bf16*)(ws + WT_BYTES + QK_BYTES);
  __bf16* vt = (__bf16*)(ws + WT_BYTES + 2 * QK_BYTES);

  (void)in_sizes; (void)n_in; (void)out_size; (void)ws_size;

  wt_transpose_kernel<<<(3 * HSd * Cdim) / 256, 256, 0, stream>>>(Wq, Wk, Wv, wt);
  qkv_proj_kernel<<<(Bsz * Tseq) / 16, 96, 0, stream>>>(x, wt, qb, kb, vt);
  flash_attn_kernel<<<Bsz * (Tseq / 64), 128, 0, stream>>>(qb, kb, vt, out);
}
